// RegionProposalNetwork_17514876634008
// MI455X (gfx1250) — matrix operand order repaired and checked by reading, not compiled, not run on hardware
//
#include <hip/hip_runtime.h>
#include <hip/hip_bf16.h>

typedef unsigned int u32;
typedef unsigned long long u64;
typedef __bf16 bf16;
typedef __attribute__((ext_vector_type(8)))  bf16  v8bf;
typedef __attribute__((ext_vector_type(16))) bf16  v16bf;
typedef __attribute__((ext_vector_type(8)))  float v8f;

#define HW     16384   // 128*128 pixels
#define NANCH  147456  // HW*9
#define NPRE   12000
#define NPOST  2000
#define NMSW   376     // u32 words per NMS matrix row (12032 bits)
#define NCAND  16384
#define KEY_NEGINF 0x007FFFFFu

// ---- d_out offsets (float elements), concat in return order ----
#define O_LOCS   0          // 147456*4
#define O_SCORES 589824     // 147456*2
#define O_ROIS   884736     // 2000*4
#define O_MASK   892736     // 2000
#define O_RIDX   894736     // 2000
#define O_ANCH   896736     // 147456*4
#define O_TOTAL  1486560

// ---- workspace offsets (bytes) ----
#define WS_HT    0ull          // hT[p][c]: 16384*512 bf16 = 16,777,216
#define WS_WB    16777216ull   // wB[r][cout][cin] bf16    =  4,718,592
#define WS_HID   21495808ull   // hid[p][c] bf16           = 16,777,216
#define WS_KEYS  38273024ull   // 147456 u32               =    589,824
#define WS_ROIS  38862848ull   // 147456*4 f32             =  2,359,296
#define WS_HIST  41222144ull   // 65536 u32                =    262,144
#define WS_HIST2 41484288ull   // 65536 u32                =    262,144
#define WS_SEL   41746432ull   // 64 u32 (sel/counter)     =        256
#define WS_CAND  41746688ull   // 16384 u64                =    131,072
#define WS_SBOX  41877760ull   // 12032*4 f32              =    192,512
#define WS_SAREA 42070272ull   // 12032 f32                =     48,128
#define WS_SVAL  42118400ull   // 12032 u32                =     48,128
#define WS_MAT   42166528ull   // 12000*376 u32            = 18,048,000
#define ZERO_U32_COUNT ((41877760ull - 41222144ull) / 4ull)  // 163,904 words

// conv LDS geometry
#define PITCH    40                 // bf16 row pitch (80B: 16B-aligned, conflict-free)
#define PAD_ROW  (130 * PITCH)      // 5200 elems per halo row
#define SPAD_ELEMS (3 * PAD_ROW)    // 15,600
#define SB_TILE  (128 * PITCH)      // 5120 elems per B tile
#define SMEM_ELEMS (SPAD_ELEMS + 9 * SB_TILE)   // 61,680 elems
#define SMEM_BYTES (SMEM_ELEMS * 2)             // 123,360 B

// ---------------------------------------------------------------------------
__global__ void k_zero_out(float* out) {
    int i = blockIdx.x * 256 + threadIdx.x;
    if (i < O_TOTAL) out[i] = 0.0f;
}
__global__ void k_zero_ws(u32* w) {
    u32 i = blockIdx.x * 256 + threadIdx.x;
    if (i < (u32)ZERO_U32_COUNT) w[i] = 0u;
}

// ---------------------------------------------------------------------------
// K1: transpose+convert h [512][16384] f32 -> hT [16384][512] bf16 (LDS-tiled)
__global__ void __launch_bounds__(256)
k_cvt_h(const float* __restrict__ h, bf16* __restrict__ hT) {
    __shared__ bf16 tile[128 * PITCH];
    const int pt = blockIdx.x & 127;          // pixel tile (128 pixels)
    const int ct = blockIdx.x >> 7;           // channel tile (32 channels)
    const int p0 = pt * 128, c0 = ct * 32;
    const int t = threadIdx.x;
    {   // load: each thread: 1 channel row, 16 consecutive pixels (4x float4)
        const int c = t & 31, pq = t >> 5;
        const float* src = h + (u64)(c0 + c) * 16384ull + p0 + pq * 16;
        const float4* s4 = (const float4*)src;
        float4 f[4] = { s4[0], s4[1], s4[2], s4[3] };
        const float* fv = (const float*)f;
#pragma unroll
        for (int e = 0; e < 16; ++e)
            tile[(pq * 16 + e) * PITCH + c] = (bf16)fv[e];
    }
    __syncthreads();
    {   // store: 2 threads per pixel, 16 channels each, 2x16B vector stores
        const int p = t >> 1, half = t & 1;
        const bf16* srcl = &tile[p * PITCH + half * 16];
        bf16* dst = hT + (u64)(p0 + p) * 512ull + c0 + half * 16;
        ((v8bf*)dst)[0] = ((const v8bf*)srcl)[0];
        ((v8bf*)dst)[1] = ((const v8bf*)srcl)[1];
    }
}

// K1b: conv1_w (cout,cin,3,3) f32 -> wB[r][cout][cin] bf16
__global__ void k_cvt_w(const float* __restrict__ w, bf16* __restrict__ wb) {
    u32 i = blockIdx.x * 256 + threadIdx.x;
    if (i >= 9u * 512u * 512u) return;
    u32 r = i / (512u * 512u);
    u32 rem = i % (512u * 512u);
    u32 n = rem / 512u;       // cout
    u32 k = rem % 512u;       // cin
    wb[i] = (bf16)w[((u64)n * 512u + k) * 9u + r];
}

// ---------------------------------------------------------------------------
// K2: 3x3 conv as WMMA implicit GEMM with zero-padded halo staging.
// grid = 512 blocks (128 rows x 4 cout-tiles), 256 threads (8 waves).
__global__ void __launch_bounds__(256)
k_conv3x3(const bf16* __restrict__ hT, const bf16* __restrict__ wb,
          const float* __restrict__ bias, bf16* __restrict__ hid) {
    extern __shared__ bf16 smem[];
    bf16* sPad = smem;                 // [3 rows][130 xpad][32 c], pitch 40
    bf16* sB   = smem + SPAD_ELEMS;    // [9 r][128 n][32 k], pitch 40

    const int mt = blockIdx.x & 127;          // image row y
    const int nbase = (blockIdx.x >> 7) * 128;
    const int t = threadIdx.x;
    const int lane = t & 31;
    const int wv = t >> 5;
    const int Mo = (wv >> 1) * 32;
    const int No = (wv & 1) * 64;
    const int g = lane >> 4;
    const int l16 = lane & 15;

    v8f acc[2][4];
#pragma unroll
    for (int mi = 0; mi < 2; ++mi)
#pragma unroll
        for (int ni = 0; ni < 4; ++ni)
            acc[mi][ni] = (v8f)0.0f;

    for (int kb = 0; kb < 16; ++kb) {
        const int cb = kb * 32;
        __syncthreads();
        // ---- stage padded A halo: 3 rows x 130 xpad x 32 ch (390 tasks) ----
#pragma unroll
        for (int s = 0; s < 2; ++s) {
            const int task = t + 256 * s;
            if (task < 390) {
                const int row = task / 130;       // 0..2 -> y-1..y+1
                const int xp = task % 130;        // 0..129 -> x = xp-1
                const int x = xp - 1;
                const int y2 = mt + row - 1;
                bf16* dst = &sPad[row * PAD_ROW + xp * PITCH];
                if ((unsigned)x < 128u && (unsigned)y2 < 128u) {
                    const v8bf* src = (const v8bf*)(hT + ((u64)(y2 * 128 + x) * 512ull + cb));
                    v8bf a0 = src[0], a1 = src[1], a2 = src[2], a3 = src[3];
                    ((v8bf*)dst)[0] = a0; ((v8bf*)dst)[1] = a1;
                    ((v8bf*)dst)[2] = a2; ((v8bf*)dst)[3] = a3;
                } else {
                    v8bf z = (v8bf)(bf16)0.0f;
                    ((v8bf*)dst)[0] = z; ((v8bf*)dst)[1] = z;
                    ((v8bf*)dst)[2] = z; ((v8bf*)dst)[3] = z;
                }
            }
        }
        // ---- stage all 9 B tiles: 9 x 128 n x 32 k (2304 tasks) ----
#pragma unroll
        for (int s = 0; s < 9; ++s) {
            const int task = t + 256 * s;
            const int r = task >> 8;
            const int rem = task & 255;
            const int n = rem & 127;
            const int half = rem >> 7;
            const bf16* src = wb + (u64)r * 262144ull + (u64)(nbase + n) * 512ull + cb + half * 16;
            v8bf w0 = ((const v8bf*)src)[0];
            v8bf w1 = ((const v8bf*)src)[1];
            bf16* dst = &sB[r * SB_TILE + n * PITCH + half * 16];
            ((v8bf*)dst)[0] = w0;
            ((v8bf*)dst)[1] = w1;
        }
        __syncthreads();

        // ---- 9 taps x 8 WMMA ----
        for (int r = 0; r < 9; ++r) {
            const int dy = r / 3, dx = r % 3;   // halo row = dy, xpad = x + dx
            union F { v16bf v; v8bf h[2]; };
            v16bf afr[2], bfr[4];
#pragma unroll
            for (int mi = 0; mi < 2; ++mi) {
                const int base = dy * PAD_ROW + (Mo + mi * 16 + l16 + dx) * PITCH;
                F u;
                u.h[0] = *(const v8bf*)&sPad[base + g * 8];
                u.h[1] = *(const v8bf*)&sPad[base + 16 + g * 8];
                afr[mi] = u.v;
            }
#pragma unroll
            for (int ni = 0; ni < 4; ++ni) {
                const int base = r * SB_TILE + (No + ni * 16 + l16) * PITCH;
                F u;
                u.h[0] = *(const v8bf*)&sB[base + g * 8];
                u.h[1] = *(const v8bf*)&sB[base + 16 + g * 8];
                bfr[ni] = u.v;
            }
#pragma unroll
            for (int mi = 0; mi < 2; ++mi)
#pragma unroll
                for (int ni = 0; ni < 4; ++ni)
                    acc[mi][ni] = __builtin_amdgcn_wmma_f32_16x16x32_bf16(
                        false, afr[mi], false, bfr[ni], (short)0, acc[mi][ni],
                        false, false);
        }
    }

    // ---- epilogue: bias + ReLU, bounce through LDS, coalesced b128 stores ----
    __syncthreads();
    bf16* sC = smem;   // [128 x][128 n], pitch 136
#pragma unroll
    for (int mi = 0; mi < 2; ++mi) {
#pragma unroll
        for (int ni = 0; ni < 4; ++ni) {
            const int nl = No + ni * 16 + l16;
            const float b = bias[nbase + nl];
#pragma unroll
            for (int v = 0; v < 8; ++v) {
                const int x = Mo + mi * 16 + g * 8 + v;
                sC[x * 136 + nl] = (bf16)fmaxf(acc[mi][ni][v] + b, 0.0f);
            }
        }
    }
    __syncthreads();
    {
        const int p = t >> 1, half = t & 1;
        const bf16* srcl = &sC[p * 136 + half * 64];
        bf16* dst = hid + (u64)(mt * 128 + p) * 512ull + nbase + half * 64;
#pragma unroll
        for (int q = 0; q < 8; ++q)
            ((v8bf*)dst)[q] = ((const v8bf*)srcl)[q];
    }
}

// ---------------------------------------------------------------------------
// K3: 1x1 convs (score: 18ch, loc: 36ch). thread = (pixel, j<54) padded to 64.
__global__ void __launch_bounds__(256)
k_conv1x1(const bf16* __restrict__ hid,
          const float* __restrict__ sw, const float* __restrict__ sb,
          const float* __restrict__ lw, const float* __restrict__ lb,
          float* __restrict__ out) {
    const u32 gidx = blockIdx.x * 256 + threadIdx.x;
    const int j = gidx & 63;
    const int p = gidx >> 6;
    if (j >= 54 || p >= HW) return;
    const float* wrow = (j < 18) ? (sw + (u64)j * 512) : (lw + (u64)(j - 18) * 512);
    const float bb = (j < 18) ? sb[j] : lb[j - 18];
    const v8bf* hp = (const v8bf*)(hid + (u64)p * 512ull);
    float acc = 0.0f;
#pragma unroll 4
    for (int k = 0; k < 64; ++k) {
        v8bf hv = hp[k];
#pragma unroll
        for (int e = 0; e < 8; ++e)
            acc = fmaf((float)hv[e], wrow[k * 8 + e], acc);
    }
    acc += bb;
    if (j < 18) out[O_SCORES + (u64)p * 18 + j] = acc;        // == (p*9+a)*2+s
    else        out[O_LOCS   + (u64)p * 36 + (j - 18)] = acc; // == (p*9+a)*4+c
}

// ---------------------------------------------------------------------------
// K4: anchors, loc2bbox, clip, validity, sortable keys
__global__ void k_proposals(float* __restrict__ out, float* __restrict__ rois,
                            u32* __restrict__ keys, const int* imh, const int* imw) {
    const u32 i = blockIdx.x * 256 + threadIdx.x;
    if (i >= NANCH) return;
    const int p = i / 9, a = i % 9;
    const int y = p >> 7, x = p & 127;
    const float sy = y * 16.0f, sx = x * 16.0f;
    const float ratios[3] = {0.5f, 1.0f, 2.0f};
    const float scales[3] = {0.5f, 1.0f, 2.0f};
    const float rr = ratios[a / 3], ss = scales[a % 3];
    const float ah = 16.0f * ss * sqrtf(rr);
    const float aw = 16.0f * ss * sqrtf(1.0f / rr);
    const float a0 = sy + 8.0f - 0.5f * ah, a1 = sx + 8.0f - 0.5f * aw;
    const float a2 = sy + 8.0f + 0.5f * ah, a3 = sx + 8.0f + 0.5f * aw;
    out[O_ANCH + (u64)i * 4 + 0] = a0;
    out[O_ANCH + (u64)i * 4 + 1] = a1;
    out[O_ANCH + (u64)i * 4 + 2] = a2;
    out[O_ANCH + (u64)i * 4 + 3] = a3;

    const float dy = out[O_LOCS + (u64)i * 4 + 0];
    const float dx = out[O_LOCS + (u64)i * 4 + 1];
    const float dh = out[O_LOCS + (u64)i * 4 + 2];
    const float dw = out[O_LOCS + (u64)i * 4 + 3];
    const float hA = a2 - a0, wA = a3 - a1;
    const float cy = a0 + 0.5f * hA, cx = a1 + 0.5f * wA;
    const float ncy = dy * hA + cy, ncx = dx * wA + cx;
    const float nh = expf(dh) * hA, nw = expf(dw) * wA;
    const float fh = (float)(*imh), fw = (float)(*imw);
    float b0 = fminf(fmaxf(ncy - 0.5f * nh, 0.0f), fh);
    float b1 = fminf(fmaxf(ncx - 0.5f * nw, 0.0f), fw);
    float b2 = fminf(fmaxf(ncy + 0.5f * nh, 0.0f), fh);
    float b3 = fminf(fmaxf(ncx + 0.5f * nw, 0.0f), fw);
    rois[(u64)i * 4 + 0] = b0; rois[(u64)i * 4 + 1] = b1;
    rois[(u64)i * 4 + 2] = b2; rois[(u64)i * 4 + 3] = b3;
    const bool valid = ((b2 - b0) >= 16.0f) && ((b3 - b1) >= 16.0f);
    const float sc = out[O_SCORES + (u64)i * 2 + 1];
    const float ms = valid ? sc : -__builtin_inff();
    u32 u = __float_as_uint(ms);
    keys[i] = (u & 0x80000000u) ? ~u : (u | 0x80000000u);
}

// ---------------------------------------------------------------------------
// K5/K7: 16-bit histograms for radix select
__global__ void k_hist_hi(const u32* __restrict__ keys, u32* __restrict__ hist) {
    const u32 i = blockIdx.x * 256 + threadIdx.x;
    if (i < NANCH) atomicAdd(&hist[keys[i] >> 16], 1u);
}
__global__ void k_hist_lo(const u32* __restrict__ keys, const u32* __restrict__ sel,
                          u32* __restrict__ hist2) {
    const u32 i = blockIdx.x * 256 + threadIdx.x;
    if (i >= NANCH) return;
    const u32 k = keys[i];
    if ((k >> 16) == sel[0]) atomicAdd(&hist2[k & 0xFFFFu], 1u);
}

// K6/K8: top-down scan of a 65536-bucket histogram to find the NPRE boundary.
__global__ void __launch_bounds__(1024)
k_select(const u32* __restrict__ hist, int phase, u32* __restrict__ sel) {
    __shared__ u32 part[1024];
    const int t = threadIdx.x;
    u32 s = 0;
    for (int q = 0; q < 64; ++q) s += hist[t * 64 + q];
    part[t] = s;
    __syncthreads();
    if (t == 0) {
        u32 cum = phase ? sel[1] : 0u;
        int fb = 0; u32 fc = cum;
        for (int seg = 1023; seg >= 0; --seg) {
            if (cum + part[seg] >= (u32)NPRE) {
                u32 c = cum;
                for (int b = seg * 64 + 63; b >= seg * 64; --b) {
                    if (c + hist[b] >= (u32)NPRE) { fb = b; fc = c; break; }
                    c += hist[b];
                }
                break;
            }
            cum += part[seg];
        }
        if (phase == 0) { sel[0] = (u32)fb; sel[1] = fc; }
        else            { sel[2] = (sel[0] << 16) | (u32)fb; }
    }
}

// K9: compact candidates >= threshold as (key<<32)|~idx
__global__ void k_compact(const u32* __restrict__ keys, const u32* __restrict__ sel,
                          u32* __restrict__ cnt, u64* __restrict__ cand) {
    const u32 i = blockIdx.x * 256 + threadIdx.x;
    if (i >= NANCH) return;
    const u32 k = keys[i];
    if (k >= sel[2]) {
        u32 pos = atomicAdd(cnt, 1u);
        if (pos < NCAND) cand[pos] = ((u64)k << 32) | (u32)(~i);
    }
}

// K10: single-WG bitonic sort of 16384 u64, descending (ties -> lower idx first)
__global__ void __launch_bounds__(1024)
k_sort(u64* __restrict__ cand) {
    const u32 tid = threadIdx.x;
    for (u32 k = 2; k <= NCAND; k <<= 1) {
        for (u32 j = k >> 1; j > 0; j >>= 1) {
            for (u32 idx = tid; idx < NCAND; idx += 1024) {
                const u32 ixj = idx ^ j;
                if (ixj > idx) {
                    u64 A = cand[idx], B = cand[ixj];
                    const bool desc = ((idx & k) == 0);
                    if (desc ? (A < B) : (A > B)) { cand[idx] = B; cand[ixj] = A; }
                }
            }
            __syncthreads();
        }
    }
}

// K10b: gather sorted top-12000 boxes, areas, validity
__global__ void k_gather(const u64* __restrict__ cand, const float* __restrict__ rois,
                         float* __restrict__ sbox, float* __restrict__ sarea,
                         u32* __restrict__ sval) {
    const u32 j = blockIdx.x * 256 + threadIdx.x;
    if (j >= NPRE) return;
    const u64 c = cand[j];
    const u32 key = (u32)(c >> 32);
    if (key == 0u) {
        sbox[(u64)j * 4 + 0] = 0; sbox[(u64)j * 4 + 1] = 0;
        sbox[(u64)j * 4 + 2] = 0; sbox[(u64)j * 4 + 3] = 0;
        sarea[j] = 1.0f; sval[j] = 0u; return;
    }
    const u32 i = ~((u32)c);
    const float b0 = rois[(u64)i * 4 + 0], b1 = rois[(u64)i * 4 + 1];
    const float b2 = rois[(u64)i * 4 + 2], b3 = rois[(u64)i * 4 + 3];
    sbox[(u64)j * 4 + 0] = b0; sbox[(u64)j * 4 + 1] = b1;
    sbox[(u64)j * 4 + 2] = b2; sbox[(u64)j * 4 + 3] = b3;
    sarea[j] = (b2 - b0 + 1.0f) * (b3 - b1 + 1.0f);
    sval[j] = (key > KEY_NEGINF) ? 1u : 0u;
}

// K11: suppression bit-matrix: mat[i][w] bit jj set if iou(i, w*32+jj) > 0.7, j>i
__global__ void __launch_bounds__(256)
k_nms_matrix(const float* __restrict__ sbox, const float* __restrict__ sarea,
             u32* __restrict__ mat) {
    const u32 t = blockIdx.x * 256 + threadIdx.x;
    if (t >= (u32)NPRE * NMSW) return;
    const u32 i = t / NMSW, w = t % NMSW;
    const float i0 = sbox[(u64)i * 4 + 0], i1 = sbox[(u64)i * 4 + 1];
    const float i2 = sbox[(u64)i * 4 + 2], i3 = sbox[(u64)i * 4 + 3];
    const float ai = sarea[i];
    u32 bits = 0u;
#pragma unroll 4
    for (u32 jj = 0; jj < 32; ++jj) {
        const u32 j = w * 32 + jj;
        if (j >= NPRE || j <= i) continue;
        const float y1 = fmaxf(i0, sbox[(u64)j * 4 + 0]);
        const float x1 = fmaxf(i1, sbox[(u64)j * 4 + 1]);
        const float y2 = fminf(i2, sbox[(u64)j * 4 + 2]);
        const float x2 = fminf(i3, sbox[(u64)j * 4 + 3]);
        const float inter = fmaxf(0.0f, y2 - y1 + 1.0f) * fmaxf(0.0f, x2 - x1 + 1.0f);
        const float iou = inter / (ai + sarea[j] - inter);
        if (iou > 0.7f) bits |= (1u << jj);
    }
    mat[(u64)i * NMSW + w] = bits;
}

// K12: sequential NMS scan + rank + scatter into d_out
__global__ void __launch_bounds__(1024)
k_nms_scan(const u32* __restrict__ mat, const u32* __restrict__ sval,
           const float* __restrict__ sbox, float* __restrict__ out) {
    __shared__ u32 removed[NMSW];
    __shared__ u32 keepbits[NMSW];
    __shared__ u32 svbits[NMSW];
    __shared__ u32 wpre[NMSW + 1];
    const int tid = threadIdx.x;
    if (tid < NMSW) {
        removed[tid] = 0u; keepbits[tid] = 0u;
        u32 b = 0u;
        for (int q = 0; q < 32; ++q) {
            const u32 j = tid * 32 + q;
            if (j < NPRE && sval[j]) b |= (1u << q);
        }
        svbits[tid] = b;
    }
    __syncthreads();
    for (int i = 0; i < NPRE; ++i) {
        const u32 w = (u32)i >> 5, b = (u32)i & 31u;
        const bool ki = ((svbits[w] >> b) & 1u) && !((removed[w] >> b) & 1u);
        if (ki) {
            if (tid < NMSW) {
                removed[tid] |= mat[(u64)i * NMSW + tid];
                if (tid == (int)w) keepbits[w] |= (1u << b);
            }
            __syncthreads();
        }
    }
    if (tid == 0) {
        u32 acc = 0;
        for (int w = 0; w < NMSW; ++w) { wpre[w] = acc; acc += __popc(keepbits[w]); }
        wpre[NMSW] = acc;
    }
    __syncthreads();
    for (int i = tid; i < NPRE; i += 1024) {
        const u32 w = (u32)i >> 5, b = (u32)i & 31u;
        if ((keepbits[w] >> b) & 1u) {
            const u32 rank = wpre[w] + __popc(keepbits[w] & ((1u << b) - 1u));
            if (rank < (u32)NPOST) {
                out[O_ROIS + (u64)rank * 4 + 0] = sbox[(u64)i * 4 + 0];
                out[O_ROIS + (u64)rank * 4 + 1] = sbox[(u64)i * 4 + 1];
                out[O_ROIS + (u64)rank * 4 + 2] = sbox[(u64)i * 4 + 2];
                out[O_ROIS + (u64)rank * 4 + 3] = sbox[(u64)i * 4 + 3];
                out[O_MASK + rank] = 1.0f;
            }
        }
    }
}

// ---------------------------------------------------------------------------
extern "C" void kernel_launch(void* const* d_in, const int* in_sizes, int n_in,
                              void* d_out, int out_size, void* d_ws, size_t ws_size,
                              hipStream_t stream) {
    const float* h       = (const float*)d_in[0];
    const float* conv1_w = (const float*)d_in[1];
    const float* conv1_b = (const float*)d_in[2];
    const float* score_w = (const float*)d_in[3];
    const float* score_b = (const float*)d_in[4];
    const float* loc_w   = (const float*)d_in[5];
    const float* loc_b   = (const float*)d_in[6];
    const int*   img_h   = (const int*)d_in[7];
    const int*   img_w   = (const int*)d_in[8];
    float* out = (float*)d_out;
    char* ws = (char*)d_ws;

    bf16* hT    = (bf16*)(ws + WS_HT);
    bf16* wb    = (bf16*)(ws + WS_WB);
    bf16* hid   = (bf16*)(ws + WS_HID);
    u32*  keys  = (u32*)(ws + WS_KEYS);
    float* rois = (float*)(ws + WS_ROIS);
    u32*  hist  = (u32*)(ws + WS_HIST);
    u32*  hist2 = (u32*)(ws + WS_HIST2);
    u32*  sel   = (u32*)(ws + WS_SEL);
    u64*  cand  = (u64*)(ws + WS_CAND);
    float* sbox = (float*)(ws + WS_SBOX);
    float* sarea= (float*)(ws + WS_SAREA);
    u32*  sval  = (u32*)(ws + WS_SVAL);
    u32*  mat   = (u32*)(ws + WS_MAT);

    hipFuncSetAttribute((const void*)k_conv3x3,
                        hipFuncAttributeMaxDynamicSharedMemorySize, SMEM_BYTES);

    k_zero_out<<<(O_TOTAL + 255) / 256, 256, 0, stream>>>(out);
    k_zero_ws<<<((int)ZERO_U32_COUNT + 255) / 256, 256, 0, stream>>>((u32*)(ws + WS_HIST));

    k_cvt_h<<<2048, 256, 0, stream>>>(h, hT);
    k_cvt_w<<<(9 * 512 * 512) / 256, 256, 0, stream>>>(conv1_w, wb);

    k_conv3x3<<<512, 256, SMEM_BYTES, stream>>>(hT, wb, conv1_b, hid);

    k_conv1x1<<<(HW * 64) / 256, 256, 0, stream>>>(hid, score_w, score_b, loc_w, loc_b, out);

    k_proposals<<<(NANCH + 255) / 256, 256, 0, stream>>>(out, rois, keys, img_h, img_w);

    k_hist_hi<<<(NANCH + 255) / 256, 256, 0, stream>>>(keys, hist);
    k_select<<<1, 1024, 0, stream>>>(hist, 0, sel);
    k_hist_lo<<<(NANCH + 255) / 256, 256, 0, stream>>>(keys, sel, hist2);
    k_select<<<1, 1024, 0, stream>>>(hist2, 1, sel);
    k_compact<<<(NANCH + 255) / 256, 256, 0, stream>>>(keys, sel, &sel[4], cand);
    k_sort<<<1, 1024, 0, stream>>>(cand);
    k_gather<<<(NPRE + 255) / 256, 256, 0, stream>>>(cand, rois, sbox, sarea, sval);
    k_nms_matrix<<<(NPRE * NMSW + 255) / 256, 256, 0, stream>>>(sbox, sarea, mat);
    k_nms_scan<<<1, 1024, 0, stream>>>(mat, sval, sbox, out);
}
